// ROIGather_69217692942862
// MI455X (gfx1250) — matrix operand order repaired and checked by reading, not compiled, not run on hardware
//
#include <hip/hip_runtime.h>
#include <stdint.h>

// ---------------------------------------------------------------- types
typedef _Float16 half_t;
typedef __attribute__((ext_vector_type(16))) _Float16 v16h;
typedef __attribute__((ext_vector_type(8)))  _Float16 v8h;
typedef __attribute__((ext_vector_type(8)))  float    v8f;

#define NB   32
#define NP   192
#define CCH  128
#define NS   36
#define MID  64
#define FCH  128
#define NROI (NB*NP)      // 6144
#define NPOS 250

__device__ __forceinline__ v8f wmma_f16(v16h a, v16h b, v8f c) {
    return __builtin_amdgcn_wmma_f32_16x16x32_f16(false, a, false, b, (short)0, c, false, false);
}

// A fragment from pre-packed global buffer: tile = mt*nKt + kt, lane reads 32B.
__device__ __forceinline__ v16h afrag_g(const half_t* base, int mt, int kt, int nKt, int lane) {
    return *(const v16h*)(base + (((size_t)(mt * nKt + kt)) << 9) + (lane << 4));
}
// B fragment: 16 consecutive halves (caller adds (lane&16)?16:0 k-offset).
__device__ __forceinline__ v16h bfrag(const half_t* p) { return *(const v16h*)p; }

// A fragment from LDS row of f16 (row-major M x K), contiguous k order (k0 + 16*(lane/16) + i), as bfrag.
__device__ __forceinline__ v16h afrag_lds_h(const half_t* row, int k0, int lane) {
    int off = k0 + ((lane & 16) ? 16 : 0);
    v8h lo = *(const v8h*)(row + off);
    v8h hi = *(const v8h*)(row + off + 8);
    v16h r;
#pragma unroll
    for (int i = 0; i < 8; i++) { r[i] = lo[i]; r[8 + i] = hi[i]; }
    return r;
}
// A fragment from LDS row of f32 (converted on the fly).
__device__ __forceinline__ v16h afrag_lds_f(const float* row, int k0, int lane) {
    int off = k0 + ((lane & 16) ? 16 : 0);
    v16h r;
#pragma unroll
    for (int i = 0; i < 8; i++) { r[i] = (half_t)row[off + i]; r[8 + i] = (half_t)row[off + 8 + i]; }
    return r;
}

// ---------------------------------------------------------------- weight repack
// mode 0: k = t*kdiv + ci, src[(m*kdiv+ci)*9 + t]   (conv9 weights)
// mode 1: src[m*K + k]                              (dense weights)
__global__ __launch_bounds__(256) void pack_a_kernel(const float* __restrict__ src,
                                                     half_t* __restrict__ dst,
                                                     int nMt, int nKt, int mode, int kdiv) {
    int total = nMt * nKt * 512;
    for (int p = blockIdx.x * 256 + threadIdx.x; p < total; p += gridDim.x * 256) {
        int tile = p >> 9;
        int t    = p & 511;
        int lane = t >> 4;
        int i    = t & 15;
        int kin  = i + ((lane & 16) ? 16 : 0);
        int mt = tile / nKt, kt = tile % nKt;
        int m = mt * 16 + (lane & 15);
        int k = kt * 32 + kin;
        float v;
        if (mode == 0) { int tt = k / kdiv, ci = k % kdiv; v = src[(m * kdiv + ci) * 9 + tt]; }
        else           { v = src[(size_t)m * (nKt * 32) + k]; }
        dst[p] = (half_t)v;
    }
}

__global__ __launch_bounds__(256) void bn_pack_kernel(const float* g, const float* b,
                                                      const float* m, const float* v,
                                                      float* sc, float* sh, int n) {
    int i = blockIdx.x * 256 + threadIdx.x;
    if (i < n) {
        float s = g[i] * rsqrtf(v[i] + 1e-5f);
        sc[i] = s;
        sh[i] = b[i] - m[i] * s;
    }
}

// ---------------------------------------------------------------- fused conv path
// One WG per 4 rois (R=4 batched through the GEMM N dimension: N = 4*36 = 144 = 9 ntiles,
// no padding waste). Stage1: 3 level convs (M=64,K=1152) -> BN+ReLU -> LDS cat buffer.
// Stage2: catconv (M=128,K=1728) -> BN+ReLU -> cat_flat[roi][c*36+s] (f16).
// Each wave register-blocks all 9 ntiles, so each packed A tile is read exactly once per WG.
#define XPITCH 144
#define CPITCH 208
#define XL_HALVES (4*3*44*XPITCH)                  // 76032 halves
#define CL_HALVES (4*44*CPITCH)                    // 36608 halves
#define CONV_LDS_BYTES ((XL_HALVES + CL_HALVES)*2) // 225280 B (dynamic LDS; 320KB/WGP on CDNA5)

__global__ __launch_bounds__(256) void convfuse_kernel(
    const float* __restrict__ x0, const float* __restrict__ x1, const float* __restrict__ x2,
    const half_t* __restrict__ packConv, const half_t* __restrict__ packCat,
    const float* __restrict__ bn1s, const float* __restrict__ bn1h,
    const float* __restrict__ bn2s, const float* __restrict__ bn2h,
    half_t* __restrict__ cat_flat) {
    extern __shared__ half_t sm[];
    half_t* xl = sm;                 // [roi][lv][row 0..43][ci]  (rows 0-3,40-43 = conv pad)
    half_t* cl = sm + XL_HALVES;     // [roi][row 0..43][cc]
    int n0 = blockIdx.x * 4;
    int tid = threadIdx.x, lane = tid & 31, w = tid >> 5;

    // zero only the padding rows
    for (int i = tid; i < 12 * 8 * XPITCH; i += 256) {
        int rl = i / (8 * XPITCH);
        int rem = i % (8 * XPITCH);
        int pr = rem / XPITCH, ci = rem % XPITCH;
        int row = pr < 4 ? pr : 36 + pr;
        xl[(rl * 44 + row) * XPITCH + ci] = (half_t)0.f;
    }
    for (int i = tid; i < 4 * 8 * CPITCH; i += 256) {
        int rr = i / (8 * CPITCH);
        int rem = i % (8 * CPITCH);
        int pr = rem / CPITCH, cc = rem % CPITCH;
        int row = pr < 4 ? pr : 36 + pr;
        cl[(rr * 44 + row) * CPITCH + cc] = (half_t)0.f;
    }

    const float* srcs[3] = {x0, x1, x2};
    for (int r = 0; r < 4; r++) {
#pragma unroll
        for (int lv = 0; lv < 3; lv++) {
            const float* s = srcs[lv] + (size_t)(n0 + r) * (CCH * NS);
            for (int i = tid; i < CCH * NS; i += 256) {
                int ci = i / NS, sp = i % NS;
                xl[((r * 3 + lv) * 44 + sp + 4) * XPITCH + ci] = (half_t)s[i];
            }
        }
    }
    __syncthreads();

    // Stage 1: 12 units (lv,mt), each with 9 ntile accumulators; K = 36 tiles of 32
    for (int unit = w; unit < 12; unit += 8) {
        int lv = unit >> 2, mt = unit & 3;
        const half_t* pa = packConv + (size_t)lv * 4 * 36 * 512;
        int base[9];
#pragma unroll
        for (int nt = 0; nt < 9; nt++) {
            int n = nt * 16 + (lane & 15);
            int roi = n / 36, sp = n % 36;
            base[nt] = ((roi * 3 + lv) * 44 + sp) * XPITCH + ((lane & 16) ? 16 : 0);
        }
        v8f acc[9];
#pragma unroll
        for (int nt = 0; nt < 9; nt++) acc[nt] = (v8f){};
        for (int kt = 0; kt < 36; kt++) {
            int t = kt >> 2;                   // k = t*128 + ci; t constant per K-tile
            int ci0 = (kt & 3) * 32;
            v16h a = afrag_g(pa, mt, kt, 36, lane);
            int off = t * XPITCH + ci0;
#pragma unroll
            for (int nt = 0; nt < 9; nt++) {
                v16h b = bfrag(xl + base[nt] + off);
                acc[nt] = wmma_f16(a, b, acc[nt]);
            }
        }
        float sc[8], sh[8];
#pragma unroll
        for (int r = 0; r < 8; r++) {
            int mo = mt * 16 + ((lane & 16) ? 8 : 0) + r;
            sc[r] = bn1s[lv * 64 + mo];
            sh[r] = bn1h[lv * 64 + mo];
        }
#pragma unroll
        for (int nt = 0; nt < 9; nt++) {
            int n = nt * 16 + (lane & 15);
            int roi = n / 36, sp = n % 36;
#pragma unroll
            for (int r = 0; r < 8; r++) {
                int mo = mt * 16 + ((lane & 16) ? 8 : 0) + r;
                float vv = acc[nt][r] * sc[r] + sh[r];
                vv = vv > 0.f ? vv : 0.f;
                cl[(roi * 44 + sp + 4) * CPITCH + lv * 64 + mo] = (half_t)vv;
            }
        }
    }
    __syncthreads();

    // Stage 2: 8 mtiles over 8 waves, 9 ntile accumulators; K = 54 tiles of 32 (k = t*192+cc)
    {
        int mt = w;
        int base[9];
#pragma unroll
        for (int nt = 0; nt < 9; nt++) {
            int n = nt * 16 + (lane & 15);
            int roi = n / 36, sp = n % 36;
            base[nt] = (roi * 44 + sp) * CPITCH + ((lane & 16) ? 16 : 0);
        }
        v8f acc[9];
#pragma unroll
        for (int nt = 0; nt < 9; nt++) acc[nt] = (v8f){};
        for (int kt = 0; kt < 54; kt++) {
            int t = kt / 6;
            int cc0 = (kt % 6) * 32;
            v16h a = afrag_g(packCat, mt, kt, 54, lane);
            int off = t * CPITCH + cc0;
#pragma unroll
            for (int nt = 0; nt < 9; nt++) {
                v16h b = bfrag(cl + base[nt] + off);
                acc[nt] = wmma_f16(a, b, acc[nt]);
            }
        }
        float sc[8], sh[8];
#pragma unroll
        for (int r = 0; r < 8; r++) {
            int co = mt * 16 + ((lane & 16) ? 8 : 0) + r;
            sc[r] = bn2s[co];
            sh[r] = bn2h[co];
        }
#pragma unroll
        for (int nt = 0; nt < 9; nt++) {
            int n = nt * 16 + (lane & 15);
            int roi = n / 36, sp = n % 36;
            half_t* dst = cat_flat + (size_t)(n0 + roi) * (CCH * NS);
#pragma unroll
            for (int r = 0; r < 8; r++) {
                int co = mt * 16 + ((lane & 16) ? 8 : 0) + r;
                float vv = acc[nt][r] * sc[r] + sh[r];
                vv = vv > 0.f ? vv : 0.f;
                dst[co * NS + sp] = (half_t)vv;
            }
        }
    }
}

// ---------------------------------------------------------------- FC + LayerNorm + ReLU
// One WG handles 64 rois: GEMM M=128(FCH) x N=64 x K=4608, then LN over FCH.
// Each wave owns one M-tile and register-blocks all 4 N-tiles -> packFc read once per WG.
#define HPITCH 132
__global__ __launch_bounds__(256) void fc_ln_kernel(
    const half_t* __restrict__ packFc, const half_t* __restrict__ cat_flat,
    const float* __restrict__ fc_b, const float* __restrict__ ln_g, const float* __restrict__ ln_b,
    float* __restrict__ roi_g) {
    __shared__ float hl[64 * HPITCH];
    __shared__ float mu_s[64], rs_s[64];
    int tid = threadIdx.x, lane = tid & 31, w = tid >> 5;
    int roiBase = blockIdx.x * 64;
    int mt = w;
    const half_t* brow[4];
#pragma unroll
    for (int nt = 0; nt < 4; nt++) {
        int nloc = nt * 16 + (lane & 15);
        brow[nt] = cat_flat + (size_t)(roiBase + nloc) * 4608 + ((lane & 16) ? 16 : 0);
    }
    v8f acc[4];
#pragma unroll
    for (int nt = 0; nt < 4; nt++) acc[nt] = (v8f){};
    for (int kt = 0; kt < 144; kt++) {
        if (kt + 16 < 144) __builtin_prefetch(brow[0] + (kt + 16) * 32, 0, 0);
        v16h a = afrag_g(packFc, mt, kt, 144, lane);
#pragma unroll
        for (int nt = 0; nt < 4; nt++) {
            v16h b = bfrag(brow[nt] + kt * 32);
            acc[nt] = wmma_f16(a, b, acc[nt]);
        }
    }
#pragma unroll
    for (int nt = 0; nt < 4; nt++) {
        int nloc = nt * 16 + (lane & 15);
#pragma unroll
        for (int r = 0; r < 8; r++) {
            int f = mt * 16 + ((lane & 16) ? 8 : 0) + r;
            hl[nloc * HPITCH + f] = acc[nt][r] + fc_b[f];
        }
    }
    __syncthreads();
    if (tid < 64) {
        float s = 0.f;
        for (int f = 0; f < FCH; f++) s += hl[tid * HPITCH + f];
        float mu = s * (1.f / FCH), v = 0.f;
        for (int f = 0; f < FCH; f++) { float d = hl[tid * HPITCH + f] - mu; v += d * d; }
        mu_s[tid] = mu;
        rs_s[tid] = rsqrtf(v * (1.f / FCH) + 1e-5f);
    }
    __syncthreads();
    for (int i = tid; i < 64 * FCH; i += 256) {
        int rr = i >> 7, f = i & 127;
        float x = (hl[rr * HPITCH + f] - mu_s[rr]) * rs_s[rr] * ln_g[f] + ln_b[f];
        roi_g[(size_t)(roiBase + rr) * FCH + f] = x > 0.f ? x : 0.f;
    }
}

// ---------------------------------------------------------------- key / value maps
// Only the 250 resized positions are computed. GEMM M=128 x N=128(pos half) x K=128.
#define FPITCH 144
__global__ __launch_bounds__(256) void keyval_kernel(
    const float* __restrict__ fmap,
    const half_t* __restrict__ packKey, const half_t* __restrict__ packVal,
    const float* __restrict__ bnks, const float* __restrict__ bnkh,
    const float* __restrict__ val_b,
    half_t* __restrict__ k_g, half_t* __restrict__ v_g) {
    __shared__ half_t fl[128 * FPITCH];
    int b = blockIdx.x >> 1, half = blockIdx.x & 1;
    int tid = threadIdx.x, lane = tid & 31, w = tid >> 5;
    int posBase = half * 128;

    for (int i = tid; i < 128 * 128; i += 256) {
        int p = i >> 7, c = i & 127;
        int gp = posBase + p;
        float v = 0.f;
        if (gp < NPOS) {
            int ii = gp / 25, jj = gp % 25;
            v = fmap[((size_t)b * CCH + c) * 4000 + (ii * 4) * 100 + jj * 4];
        }
        fl[p * FPITCH + c] = (half_t)v;
    }
    __syncthreads();

    for (int tile = w; tile < 128; tile += 8) {
        int mat = tile >> 6, rem = tile & 63, mt = rem >> 3, nt = rem & 7;
        const half_t* pk = mat ? packVal : packKey;
        int nloc = nt * 16 + (lane & 15);
        const half_t* brow = fl + nloc * FPITCH + ((lane & 16) ? 16 : 0);
        v8f acc = {};
#pragma unroll
        for (int kt = 0; kt < 4; kt++) {
            v16h a = afrag_g(pk, mt, kt, 4, lane);
            v16h bf = bfrag(brow + kt * 32);
            acc = wmma_f16(a, bf, acc);
        }
        int gp = posBase + nloc;
        if (gp < NPOS) {
            int o0 = mt * 16 + ((lane & 16) ? 8 : 0);
            if (mat == 0) {
                v8h tmp;
#pragma unroll
                for (int r = 0; r < 8; r++) {
                    float vv = acc[r] * bnks[o0 + r] + bnkh[o0 + r];
                    tmp[r] = (half_t)(vv > 0.f ? vv : 0.f);
                }
                *(v8h*)(k_g + ((size_t)b * 256 + gp) * CCH + o0) = tmp;
            } else {
#pragma unroll
                for (int r = 0; r < 8; r++) {
                    float vv = acc[r] + val_b[o0 + r];
                    v_g[((size_t)b * CCH + (o0 + r)) * 256 + gp] = (half_t)vv;
                }
            }
        }
    }
    if (half == 1) { // zero padded positions 250..255
        for (int i = tid; i < 6 * CCH; i += 256) {
            int pp = 250 + i / CCH, c = i % CCH;
            k_g[((size_t)b * 256 + pp) * CCH + c] = (half_t)0.f;
            v_g[((size_t)b * CCH + c) * 256 + pp] = (half_t)0.f;
        }
    }
}

// ---------------------------------------------------------------- attention
// One WG per (b, 48-prior chunk): q@k (48x256, K=128), softmax, sim@v (48x128, K=256).
#define QPITCH 144
#define SPITCH 264
__global__ __launch_bounds__(256) void attn_kernel(
    const float* __restrict__ roi_g, const half_t* __restrict__ k_g, const half_t* __restrict__ v_g,
    const float* __restrict__ q_w, const float* __restrict__ q_b,
    const float* __restrict__ gate_w, const float* __restrict__ gate_b,
    float* __restrict__ out) {
    __shared__ half_t ql[48 * QPITCH];
    __shared__ float  sl[48 * SPITCH];
    int b = blockIdx.x >> 2, chunk = blockIdx.x & 3;
    int npBase = chunk * 48;
    int tid = threadIdx.x, lane = tid & 31, w = tid >> 5;

    for (int i = tid; i < 48 * FCH; i += 256) {
        int r = i >> 7, c = i & 127;
        int np = npBase + r;
        float x = roi_g[((size_t)b * NP + np) * FCH + c];
        float qq = x * q_w[np] + q_b[np];
        ql[r * QPITCH + c] = (half_t)(qq > 0.f ? qq : 0.f);
    }
    __syncthreads();

    const float scale = 0.08838834764831845f; // 128^-0.5
    // GEMM1: 3 mtiles x 16 ntiles; wave w owns ntiles {2w, 2w+1}
#pragma unroll
    for (int ti = 0; ti < 6; ti++) {
        int nt = 2 * w + (ti & 1), mt = ti >> 1;
        int nloc = nt * 16 + (lane & 15);                    // pos
        const half_t* brow = k_g + ((size_t)b * 256 + nloc) * CCH + ((lane & 16) ? 16 : 0);
        const half_t* arow = ql + (mt * 16 + (lane & 15)) * QPITCH;
        v8f acc = {};
#pragma unroll
        for (int kt = 0; kt < 4; kt++) {
            v16h a = afrag_lds_h(arow, kt * 32, lane);
            v16h bf = bfrag(brow + kt * 32);
            acc = wmma_f16(a, bf, acc);
        }
#pragma unroll
        for (int r = 0; r < 8; r++) {
            int m = mt * 16 + ((lane & 16) ? 8 : 0) + r;
            sl[m * SPITCH + nloc] = acc[r] * scale;
        }
    }
    __syncthreads();

    if (tid < 48) { // one thread per row softmax over 250 valid positions
        float* row = sl + tid * SPITCH;
        float mx = -1e30f;
        for (int p = 0; p < NPOS; p++) mx = fmaxf(mx, row[p]);
        float s = 0.f;
        for (int p = 0; p < NPOS; p++) { float e = __expf(row[p] - mx); row[p] = e; s += e; }
        float inv = 1.f / s;
        for (int p = 0; p < NPOS; p++) row[p] *= inv;
        for (int p = NPOS; p < 256; p++) row[p] = 0.f;
    }
    __syncthreads();

    // GEMM2: 3 mtiles x 8 ntiles; wave w owns ntile w
#pragma unroll
    for (int mt = 0; mt < 3; mt++) {
        int nloc = w * 16 + (lane & 15);                     // channel
        const half_t* brow = v_g + ((size_t)b * CCH + nloc) * 256 + ((lane & 16) ? 16 : 0);
        const float* arow = sl + (mt * 16 + (lane & 15)) * SPITCH;
        v8f acc = {};
#pragma unroll
        for (int kt = 0; kt < 8; kt++) {
            v16h a = afrag_lds_f(arow, kt * 32, lane);
            v16h bf = bfrag(brow + kt * 32);
            acc = wmma_f16(a, bf, acc);
        }
#pragma unroll
        for (int r = 0; r < 8; r++) {
            int np = npBase + mt * 16 + ((lane & 16) ? 8 : 0) + r;
            float ctx = acc[r] * gate_w[np] + gate_b[np];
            size_t idx = ((size_t)b * NP + np) * FCH + nloc;
            out[idx] = roi_g[idx] + ctx;
        }
    }
}

// ---------------------------------------------------------------- host
extern "C" void kernel_launch(void* const* d_in, const int* in_sizes, int n_in,
                              void* d_out, int out_size, void* d_ws, size_t ws_size,
                              hipStream_t stream) {
    (void)in_sizes; (void)n_in; (void)out_size; (void)ws_size;
    const float* roi0   = (const float*)d_in[0];
    const float* roi1   = (const float*)d_in[1];
    const float* roi2   = (const float*)d_in[2];
    const float* fmap   = (const float*)d_in[3];
    const float* conv_w = (const float*)d_in[4];
    const float* conv_g = (const float*)d_in[5];
    const float* conv_b = (const float*)d_in[6];
    const float* conv_m = (const float*)d_in[7];
    const float* conv_v = (const float*)d_in[8];
    const float* cat_w  = (const float*)d_in[9];
    const float* cat_g  = (const float*)d_in[10];
    const float* cat_b  = (const float*)d_in[11];
    const float* cat_m  = (const float*)d_in[12];
    const float* cat_v  = (const float*)d_in[13];
    const float* fc_w   = (const float*)d_in[14];
    const float* fc_b   = (const float*)d_in[15];
    const float* ln_g   = (const float*)d_in[16];
    const float* ln_b   = (const float*)d_in[17];
    const float* key_w  = (const float*)d_in[18];
    const float* key_g  = (const float*)d_in[19];
    const float* key_be = (const float*)d_in[20];
    const float* key_m  = (const float*)d_in[21];
    const float* key_v  = (const float*)d_in[22];
    const float* q_w    = (const float*)d_in[23];
    const float* q_b    = (const float*)d_in[24];
    const float* val_w  = (const float*)d_in[25];
    const float* val_b  = (const float*)d_in[26];
    const float* gate_w = (const float*)d_in[27];
    const float* gate_b = (const float*)d_in[28];
    float* out = (float*)d_out;

    uint8_t* ws = (uint8_t*)d_ws;
    half_t* packConv = (half_t*)(ws + 0);          //   442368 B (3 x 4x36 tiles)
    half_t* packCat  = (half_t*)(ws + 442368);     //   442368 B
    half_t* packFc   = (half_t*)(ws + 884736);     //  1179648 B
    half_t* packKey  = (half_t*)(ws + 2064384);    //    32768 B
    half_t* packVal  = (half_t*)(ws + 2097152);    //    32768 B
    float*  bn1s = (float*)(ws + 2129920);
    float*  bn1h = bn1s + 192;
    float*  bn2s = bn1h + 192;
    float*  bn2h = bn2s + 128;
    float*  bnks = bn2h + 128;
    float*  bnkh = bnks + 128;
    half_t* cat_flat = (half_t*)(ws + 4194304);    // 6144 x 4608 f16 = 56623104 B
    float*  roi_g    = (float*)(ws + 60817408);    // 6144 x 128 f32  =  3145728 B
    half_t* k_g      = (half_t*)(ws + 63963136);   // 32 x 256 x 128 f16
    half_t* v_g      = (half_t*)(ws + 66060288);   // 32 x 128 x 256 f16

    // weight repack into WMMA A-fragment layout
    for (int lv = 0; lv < 3; lv++)
        pack_a_kernel<<<288, 256, 0, stream>>>(conv_w + (size_t)lv * 64 * 128 * 9,
                                               packConv + (size_t)lv * 4 * 36 * 512, 4, 36, 0, 128);
    pack_a_kernel<<<864, 256, 0, stream>>>(cat_w, packCat, 8, 54, 0, 192);
    pack_a_kernel<<<2304, 256, 0, stream>>>(fc_w, packFc, 8, 144, 1, 0);
    pack_a_kernel<<<64, 256, 0, stream>>>(key_w, packKey, 8, 4, 1, 0);
    pack_a_kernel<<<64, 256, 0, stream>>>(val_w, packVal, 8, 4, 1, 0);
    bn_pack_kernel<<<1, 256, 0, stream>>>(conv_g, conv_b, conv_m, conv_v, bn1s, bn1h, 192);
    bn_pack_kernel<<<1, 256, 0, stream>>>(cat_g, cat_b, cat_m, cat_v, bn2s, bn2h, 128);
    bn_pack_kernel<<<1, 256, 0, stream>>>(key_g, key_be, key_m, key_v, bnks, bnkh, 128);

    convfuse_kernel<<<NROI / 4, 256, CONV_LDS_BYTES, stream>>>(
        roi0, roi1, roi2, packConv, packCat, bn1s, bn1h, bn2s, bn2h, cat_flat);
    fc_ln_kernel<<<NROI / 64, 256, 0, stream>>>(packFc, cat_flat, fc_b, ln_g, ln_b, roi_g);
    keyval_kernel<<<NB * 2, 256, 0, stream>>>(fmap, packKey, packVal, bnks, bnkh, val_b, k_g, v_g);
    attn_kernel<<<NB * 4, 256, 0, stream>>>(roi_g, k_g, v_g, q_w, q_b, gate_w, gate_b, out);
}
